// Biattention_Generator_16741782520559
// MI455X (gfx1250) — matrix operand order repaired and checked by reading, not compiled, not run on hardware
//
#include <hip/hip_runtime.h>
#include <hip/hip_bf16.h>
#include <math.h>

#define B_   32
#define S_   128
#define H_   1024
#define T_   32
#define E_   512
#define V_   32000
#define D_   256
#define TM1  31

typedef __bf16 bf16;
typedef __attribute__((ext_vector_type(16))) __bf16 v16bf;
typedef __attribute__((ext_vector_type(8)))  __bf16 v8bf;
typedef __attribute__((ext_vector_type(8)))  float  v8f;

__device__ __forceinline__ float bf2f(bf16 b) {
  unsigned short s = __builtin_bit_cast(unsigned short, b);
  unsigned u = ((unsigned)s) << 16;
  return __builtin_bit_cast(float, u);
}
__device__ __forceinline__ bf16 f2bf(float x) {
  unsigned u = __builtin_bit_cast(unsigned, x);
  unsigned r = u + 0x7FFFu + ((u >> 16) & 1u);
  unsigned short s = (unsigned short)(r >> 16);
  return __builtin_bit_cast(bf16, s);
}

// ---------------- conversion / packing kernels ----------------

__global__ void k_f32_to_bf16(const float* __restrict__ src, bf16* __restrict__ dst, int n) {
  int i = blockIdx.x * blockDim.x + threadIdx.x;
  if (i < n) dst[i] = f2bf(src[i]);
}

// xs_b[(t*B+b)*E + e] = gt[(b*T+t)*E + e]  (transpose (B,T,E) -> (T-1,B,E), bf16)
__global__ void k_xs_pack(const float* __restrict__ gt, bf16* __restrict__ xs) {
  int i = blockIdx.x * blockDim.x + threadIdx.x;
  if (i >= TM1 * B_ * E_) return;
  int e = i % E_;
  int r = i / E_;
  int b = r % B_;
  int t = r / B_;
  xs[i] = f2bf(gt[((size_t)b * T_ + t) * E_ + e]);
}

__global__ void k_h_init(const float* __restrict__ hv, float* __restrict__ h,
                         bf16* __restrict__ hb, int n) {
  int i = blockIdx.x * blockDim.x + threadIdx.x;
  if (i < n) { float v = hv[i]; h[i] = v; hb[i] = f2bf(v); }
}

// ---------------- generic WMMA bf16 GEMM:  C = act(A @ Bw^T + bias) * postmul ----------------
// A: (M,K) row-major bf16, lda elements.  Bw: (N,K) row-major bf16, ldb elements.
// Each wave owns one 16-wide n-tile and computes a 32x16 output strip (two m-tiles),
// reusing every B fetch across two WMMAs (halves B traffic vs one m-tile per wave).
// grid = (ceil(N/16/8), M/32); blockDim = 256 (8 waves). M % 32 == 0, N % 16 == 0, K % 32 == 0.

#define GWAVES 8
__launch_bounds__(GWAVES * 32)
__global__ void k_gemm_bf16(const bf16* __restrict__ A, int lda,
                            const bf16* __restrict__ Bw, int ldb,
                            const float* __restrict__ bias,
                            const float* __restrict__ postmul,
                            float* __restrict__ Cout, int ldc,
                            bf16*  __restrict__ CoutB, int ldcb,
                            int N, int K, int act) {
  int wave = threadIdx.x >> 5;
  int lane = threadIdx.x & 31;
  int ntile = blockIdx.x * GWAVES + wave;
  if (ntile * 16 >= N) return;                 // wave-uniform: EXEC stays all-ones
  int m0 = blockIdx.y * 32;
  int n0 = ntile * 16;
  int lh = lane >> 4;                          // lane half (0/1)
  int lm = lane & 15;

  v8f acc0, acc1;
#pragma unroll
  for (int r = 0; r < 8; ++r) { acc0[r] = 0.0f; acc1[r] = 0.0f; }

  // A layout (16-bit A 16x32): elems 0..7 -> K = k0 + lh*8 + i ; elems 8..15 -> K = k0+16+lh*8+i
  const bf16* arow0 = A + (size_t)(m0 + lm) * lda + lh * 8;
  const bf16* arow1 = arow0 + (size_t)16 * lda;
  // B layout (16-bit B 32x16): elems 0..7 -> K = k0 + lh*8 + i ; elems 8..15 -> K = k0+16+lh*8+i ; N = n0+lm  (Bw row-major (N,K))
  const bf16* brow = Bw + (size_t)(n0 + lm) * ldb + lh * 8;

  for (int k0 = 0; k0 < K; k0 += 32) {
    v8bf b0 = *(const v8bf*)(brow + k0);
    v8bf b1 = *(const v8bf*)(brow + k0 + 16);
    v8bf p0 = *(const v8bf*)(arow0 + k0);
    v8bf p1 = *(const v8bf*)(arow0 + k0 + 16);
    v8bf q0 = *(const v8bf*)(arow1 + k0);
    v8bf q1 = *(const v8bf*)(arow1 + k0 + 16);
    v16bf a0, a1, b;
#pragma unroll
    for (int i = 0; i < 8; ++i) {
      b[i] = b0[i]; b[i + 8] = b1[i];
      a0[i] = p0[i]; a0[i + 8] = p1[i];
      a1[i] = q0[i]; a1[i + 8] = q1[i];
    }
    acc0 = __builtin_amdgcn_wmma_f32_16x16x32_bf16(false, a0, false, b, (short)0, acc0, false, false);
    acc1 = __builtin_amdgcn_wmma_f32_16x16x32_bf16(false, a1, false, b, (short)0, acc1, false, false);
  }

  float bsv = bias    ? bias[n0 + lm]    : 0.0f;
  float pmv = postmul ? postmul[n0 + lm] : 1.0f;
#pragma unroll
  for (int r = 0; r < 8; ++r) {
    // C layout: VGPR r -> M = r + 8*half, N = lane&15
    float v0 = acc0[r] + bsv;
    float v1 = acc1[r] + bsv;
    if (act == 1) { v0 = tanhf(v0); v1 = tanhf(v1); }
    v0 *= pmv; v1 *= pmv;
    int ma = m0 + lh * 8 + r;
    int mb = ma + 16;
    if (Cout) {
      Cout[(size_t)ma * ldc + n0 + lm] = v0;
      Cout[(size_t)mb * ldc + n0 + lm] = v1;
    }
    if (CoutB) {
      CoutB[(size_t)ma * ldcb + n0 + lm] = f2bf(v0);
      CoutB[(size_t)mb * ldcb + n0 + lm] = f2bf(v1);
    }
  }
}

// ---------------- per-step attention: scores -> softmax -> mix ----------------
// one block per batch row b; 128 threads (= S)

__launch_bounds__(128)
__global__ void k_attn(const bf16* __restrict__ uhp,   // (B,D) bf16 = tanh(hU^T+Ub)*P_w
                       const bf16* __restrict__ vf,    // (B*S,D) bf16 = tanh(fV^T+Vb)
                       const float* __restrict__ f,    // (B,S,H) fp32
                       const float* __restrict__ Pb,
                       float* __restrict__ mix, bf16* __restrict__ mixb) {
  __shared__ float red[128];
  __shared__ float satt[128];
  int b = blockIdx.x, s = threadIdx.x;

  const bf16* u  = uhp + b * D_;
  const bf16* vv = vf + ((size_t)b * S_ + s) * D_;
  float dot = Pb[0];
#pragma unroll 4
  for (int d = 0; d < D_; ++d) dot += bf2f(u[d]) * bf2f(vv[d]);

  red[s] = dot; __syncthreads();
  for (int st = 64; st > 0; st >>= 1) { if (s < st) red[s] = fmaxf(red[s], red[s + st]); __syncthreads(); }
  float mx = red[0]; __syncthreads();
  float e = __expf(dot - mx);
  red[s] = e; __syncthreads();
  for (int st = 64; st > 0; st >>= 1) { if (s < st) red[s] += red[s + st]; __syncthreads(); }
  satt[s] = e / red[0]; __syncthreads();

  float acc[8];
#pragma unroll
  for (int j = 0; j < 8; ++j) acc[j] = 0.0f;
  for (int s2 = 0; s2 < S_; ++s2) {
    float a = satt[s2];
    const float* fr = f + ((size_t)b * S_ + s2) * H_ + s;
#pragma unroll
    for (int j = 0; j < 8; ++j) acc[j] += a * fr[j * 128];
  }
#pragma unroll
  for (int j = 0; j < 8; ++j) {
    int h = s + j * 128;
    mix [b * H_ + h] = acc[j];
    mixb[b * H_ + h] = f2bf(acc[j]);
  }
}

// ---------------- GRU elementwise combine ----------------

__global__ void k_gru(const float* __restrict__ GXt,   // (B,3H) = x@Wih_x^T (this step)
                      const float* __restrict__ GMIX,  // (B,3H) = mix@Wih_h^T
                      const float* __restrict__ GH,    // (B,3H) = h@Whh^T
                      const float* __restrict__ bih, const float* __restrict__ bhh,
                      const float* __restrict__ hcur,
                      float* __restrict__ hnext, bf16* __restrict__ hnextb) {
  int i = blockIdx.x * blockDim.x + threadIdx.x;
  if (i >= B_ * H_) return;
  int m = i >> 10, j = i & (H_ - 1);
  int base = m * 3 * H_;
  float xr = GXt[base + j]          + GMIX[base + j]          + bih[j];
  float xz = GXt[base + H_ + j]     + GMIX[base + H_ + j]     + bih[H_ + j];
  float xn = GXt[base + 2 * H_ + j] + GMIX[base + 2 * H_ + j] + bih[2 * H_ + j];
  float hr = GH[base + j]           + bhh[j];
  float hz = GH[base + H_ + j]      + bhh[H_ + j];
  float hn = GH[base + 2 * H_ + j]  + bhh[2 * H_ + j];
  float r = 1.0f / (1.0f + __expf(-(xr + hr)));
  float z = 1.0f / (1.0f + __expf(-(xz + hz)));
  float n = tanhf(xn + r * hn);
  float o = (1.0f - z) * n + z * hcur[i];
  hnext[i]  = o;
  hnextb[i] = f2bf(o);
}

// ---------------- host orchestration ----------------

extern "C" void kernel_launch(void* const* d_in, const int* in_sizes, int n_in,
                              void* d_out, int out_size, void* d_ws, size_t ws_size,
                              hipStream_t stream) {
  const float* f    = (const float*)d_in[0];
  const float* hv   = (const float*)d_in[1];
  const float* gt   = (const float*)d_in[2];
  const float* U_w  = (const float*)d_in[4];
  const float* U_b  = (const float*)d_in[5];
  const float* V_w  = (const float*)d_in[6];
  const float* V_b  = (const float*)d_in[7];
  const float* P_w  = (const float*)d_in[8];
  const float* P_b  = (const float*)d_in[9];
  const float* W_ih = (const float*)d_in[10];
  const float* b_ih = (const float*)d_in[11];
  const float* W_hh = (const float*)d_in[12];
  const float* b_hh = (const float*)d_in[13];
  const float* Dw   = (const float*)d_in[14];
  const float* Db   = (const float*)d_in[15];
  float* out = (float*)d_out;

  char* ws = (char*)d_ws;
  size_t off = 0;
  auto alloc = [&](size_t bytes) -> char* {
    char* p = ws + off; off += (bytes + 255) & ~(size_t)255; return p;
  };

  bf16*  Dw_b  = (bf16*) alloc((size_t)V_ * H_ * 2);
  bf16*  Wih_b = (bf16*) alloc((size_t)3 * H_ * (E_ + H_) * 2);
  bf16*  Whh_b = (bf16*) alloc((size_t)3 * H_ * H_ * 2);
  bf16*  Uw_b  = (bf16*) alloc((size_t)D_ * H_ * 2);
  bf16*  Vw_b  = (bf16*) alloc((size_t)D_ * H_ * 2);
  bf16*  f_b   = (bf16*) alloc((size_t)B_ * S_ * H_ * 2);
  bf16*  xs_b  = (bf16*) alloc((size_t)TM1 * B_ * E_ * 2);
  bf16*  vf_b  = (bf16*) alloc((size_t)B_ * S_ * D_ * 2);
  float* GX    = (float*)alloc((size_t)TM1 * B_ * 3 * H_ * 4);
  bf16*  uhp_b = (bf16*) alloc((size_t)B_ * D_ * 2);
  float* mix   = (float*)alloc((size_t)B_ * H_ * 4);
  bf16*  mix_b = (bf16*) alloc((size_t)B_ * H_ * 2);
  float* GMIX  = (float*)alloc((size_t)B_ * 3 * H_ * 4);
  float* GH    = (float*)alloc((size_t)B_ * 3 * H_ * 4);
  float* h0    = (float*)alloc((size_t)B_ * H_ * 4);
  bf16*  h0_b  = (bf16*) alloc((size_t)B_ * H_ * 2);
  float* h1    = (float*)alloc((size_t)B_ * H_ * 4);
  bf16*  h1_b  = (bf16*) alloc((size_t)B_ * H_ * 2);

  auto cvt = [&](const float* s, bf16* d, int n) {
    k_f32_to_bf16<<<(n + 255) / 256, 256, 0, stream>>>(s, d, n);
  };
  cvt(Dw,   Dw_b,  V_ * H_);
  cvt(W_ih, Wih_b, 3 * H_ * (E_ + H_));
  cvt(W_hh, Whh_b, 3 * H_ * H_);
  cvt(U_w,  Uw_b,  D_ * H_);
  cvt(V_w,  Vw_b,  D_ * H_);
  cvt(f,    f_b,   B_ * S_ * H_);
  { int n = TM1 * B_ * E_; k_xs_pack<<<(n + 255) / 256, 256, 0, stream>>>(gt, xs_b); }
  k_h_init<<<(B_ * H_ + 255) / 256, 256, 0, stream>>>(hv, h0, h0_b, B_ * H_);

  // vf = tanh(f @ V_w^T + V_b)  (M=4096, N=256, K=1024) -> bf16
  k_gemm_bf16<<<dim3(2, (B_ * S_) / 32), 256, 0, stream>>>(
      f_b, H_, Vw_b, H_, V_b, nullptr, nullptr, 0, vf_b, D_, D_, H_, 1);

  // GX = xs @ W_ih[:, :E]^T  (M=992, N=3072, K=512), loop-invariant precompute
  k_gemm_bf16<<<dim3(24, (TM1 * B_) / 32), 256, 0, stream>>>(
      xs_b, E_, Wih_b, E_ + H_, nullptr, nullptr, GX, 3 * H_, nullptr, 0, 3 * H_, E_, 0);

  float* hc = h0; bf16* hcb = h0_b;
  float* hn = h1; bf16* hnb = h1_b;
  for (int t = 0; t < TM1; ++t) {
    // uhp = tanh(h @ U_w^T + U_b) * P_w   (M=32, N=256, K=1024)
    k_gemm_bf16<<<dim3(2, 1), 256, 0, stream>>>(
        hcb, H_, Uw_b, H_, U_b, P_w, nullptr, 0, uhp_b, D_, D_, H_, 1);
    // attention scores + softmax + mix
    k_attn<<<B_, 128, 0, stream>>>(uhp_b, vf_b, f, P_b, mix, mix_b);
    // GMIX = mix @ W_ih[:, E:]^T   (M=32, N=3072, K=1024)
    k_gemm_bf16<<<dim3(24, 1), 256, 0, stream>>>(
        mix_b, H_, Wih_b + E_, E_ + H_, nullptr, nullptr, GMIX, 3 * H_, nullptr, 0, 3 * H_, H_, 0);
    // GH = h @ W_hh^T   (M=32, N=3072, K=1024)
    k_gemm_bf16<<<dim3(24, 1), 256, 0, stream>>>(
        hcb, H_, Whh_b, H_, nullptr, nullptr, GH, 3 * H_, nullptr, 0, 3 * H_, H_, 0);
    // GRU combine -> h_next
    k_gru<<<(B_ * H_ + 255) / 256, 256, 0, stream>>>(
        GX + (size_t)t * B_ * 3 * H_, GMIX, GH, b_ih, b_hh, hc, hn, hnb);
    // logits = h_next @ dense_w^T + dense_b   (M=32, N=32000, K=1024), out[b, t, :]
    k_gemm_bf16<<<dim3(V_ / 16 / GWAVES, 1), 256, 0, stream>>>(
        hnb, H_, Dw_b, H_, Db, nullptr, out + (size_t)t * V_, TM1 * V_, nullptr, 0, V_, H_, 0);

    float* tf = hc; hc = hn; hn = tf;
    bf16* tb = hcb; hcb = hnb; hnb = tb;
  }
  (void)in_sizes; (void)n_in; (void)out_size; (void)ws_size;
}
